// WindowAttention_90348932038789
// MI455X (gfx1250) — matrix operand order repaired and checked by reading, not compiled, not run on hardware
//
#include <hip/hip_runtime.h>
#include <hip/hip_bf16.h>

// ---------------------------------------------------------------------------
// Window attention (Swin/CSwin + LePE) for gfx1250, f16 WMMA pipeline with
// f16 intermediates and GLOBAL_LOAD_ASYNC_TO_LDS staging.
//   shapes: B_=4096 windows, N=64 tokens, C=256, HEADS=8, hd=32
// ---------------------------------------------------------------------------

typedef __attribute__((ext_vector_type(16))) _Float16 v16h;
typedef __attribute__((ext_vector_type(8)))  float    v8f;

#define WMMA_F16(a, b, c) \
    __builtin_amdgcn_wmma_f32_16x16x32_f16(false, (a), false, (b), (short)0, (c), false, false)

// --- gfx1250 async global->LDS copy (raw 16B per lane, ASYNCcnt-tracked) ----
__device__ __forceinline__ void async_ld16(unsigned lds_off, const void* g) {
    asm volatile("global_load_async_to_lds_b128 %0, %1, off"
                 :: "v"(lds_off), "v"(g) : "memory");
}
__device__ __forceinline__ void async_wait0() {
    asm volatile("s_wait_asynccnt 0x0" ::: "memory");
}
__device__ __forceinline__ unsigned lds_addr(const void* p) {
    // generic LDS pointer: aperture in high bits, LDS byte offset in [31:0]
    return (unsigned)(unsigned long long)p;
}

// --- WMMA operand gathers from LDS (wave32; contiguous k = 16*hi + i, same order in A and B) -------
// A (16x32 f16): lane row = lane&15; halves i -> K = 16*hi + i   (hi = lane>>4)
__device__ __forceinline__ v16h lds_load_a(const _Float16* Ls, int rowBase,
                                           int stride, int kOff, int lo, int hi) {
    v16h a;
    const int base = (rowBase + lo) * stride + kOff + 16 * hi;
#pragma unroll
    for (int i = 0; i < 16; ++i) a[i] = Ls[base + i];
    return a;
}

// B (32x16 f16), source stored [n][k]: lane col = lane&15; halves i -> K=16*hi+i
__device__ __forceinline__ v16h lds_load_b(const _Float16* Ls, int colBase,
                                           int stride, int kOff, int lo, int hi) {
    v16h b;
    const int base = (colBase + lo) * stride + kOff + 16 * hi;
#pragma unroll
    for (int i = 0; i < 16; ++i) b[i] = Ls[base + i];
    return b;
}

// B (32x16 f16), source stored [k][n] (un-transposed): strided gather
__device__ __forceinline__ v16h lds_load_b_rowmajor(const _Float16* Ls, int colBase,
                                                    int stride, int kOff, int lo, int hi) {
    v16h b;
    const int col = colBase + lo;
    const int kb  = kOff + 16 * hi;
#pragma unroll
    for (int i = 0; i < 16; ++i) b[i] = Ls[(kb + i) * stride + col];
    return b;
}

// ---------------------------------------------------------------------------
// Projection GEMM (f32 in -> f16 out):  Out[M,256] = In[M,256] @ W^T + bias
// Block 256 threads (8 waves). Tile BM=128, BN=64, BK=32; f32->f16 staged
// through LDS, padded stride 40 halves (80B rows -> conflict-free b128 reads).
// ---------------------------------------------------------------------------
__global__ void __launch_bounds__(256)
gemm_f32in_f16out(const float* __restrict__ A, const float* __restrict__ W,
                  const float* __restrict__ bias, _Float16* __restrict__ Out) {
    __shared__ _Float16 As[128 * 40];
    __shared__ _Float16 Bs[64 * 40];

    const int m0 = blockIdx.x * 128, n0 = blockIdx.y * 64;
    const int tid = threadIdx.x;
    const int wave = tid >> 5, lane = tid & 31, hi = lane >> 4, lo = lane & 15;
    const int rw = wave * 16;

    v8f acc[4] = {v8f{}, v8f{}, v8f{}, v8f{}};

    for (int ks = 0; ks < 256; ks += 32) {
        if (ks + 32 < 256) {
            const int c = tid, row = c >> 3, k4 = (c & 7) << 2;
            __builtin_prefetch(A + (size_t)(m0 + row) * 256 + ks + 32 + k4, 0, 0);
        }
#pragma unroll
        for (int j = 0; j < 4; ++j) {              // A tile 128x32 f32->f16
            const int c = tid + 256 * j;
            const int row = c >> 3, k4 = (c & 7) << 2;
            const float4 f = *(const float4*)(A + (size_t)(m0 + row) * 256 + ks + k4);
            As[row * 40 + k4 + 0] = (_Float16)f.x;
            As[row * 40 + k4 + 1] = (_Float16)f.y;
            As[row * 40 + k4 + 2] = (_Float16)f.z;
            As[row * 40 + k4 + 3] = (_Float16)f.w;
        }
#pragma unroll
        for (int j = 0; j < 2; ++j) {              // W tile 64x32 (B[k][n]=W[n][k])
            const int c = tid + 256 * j;
            const int row = c >> 3, k4 = (c & 7) << 2;
            const float4 f = *(const float4*)(W + (size_t)(n0 + row) * 256 + ks + k4);
            Bs[row * 40 + k4 + 0] = (_Float16)f.x;
            Bs[row * 40 + k4 + 1] = (_Float16)f.y;
            Bs[row * 40 + k4 + 2] = (_Float16)f.z;
            Bs[row * 40 + k4 + 3] = (_Float16)f.w;
        }
        __syncthreads();
        const v16h a = lds_load_a(As, rw, 40, 0, lo, hi);
#pragma unroll
        for (int t = 0; t < 4; ++t) {
            const v16h b = lds_load_b(Bs, 16 * t, 40, 0, lo, hi);
            acc[t] = WMMA_F16(a, b, acc[t]);
        }
        __syncthreads();
    }
#pragma unroll
    for (int t = 0; t < 4; ++t) {
        const int col = n0 + 16 * t + lo;
        const float bcol = bias[col];
#pragma unroll
        for (int r = 0; r < 8; ++r) {
            const int row = m0 + rw + r + 8 * hi;
            Out[(size_t)row * 256 + col] = (_Float16)(acc[t][r] + bcol);
        }
    }
}

// ---------------------------------------------------------------------------
// Output GEMM (f16 in -> f32 out): A tiles staged via async global->LDS copy.
// ---------------------------------------------------------------------------
__global__ void __launch_bounds__(256)
gemm_f16in_f32out(const _Float16* __restrict__ A, const float* __restrict__ W,
                  const float* __restrict__ bias, float* __restrict__ Out) {
    __shared__ _Float16 As[128 * 40];
    __shared__ _Float16 Bs[64 * 40];

    const int m0 = blockIdx.x * 128, n0 = blockIdx.y * 64;
    const int tid = threadIdx.x;
    const int wave = tid >> 5, lane = tid & 31, hi = lane >> 4, lo = lane & 15;
    const int rw = wave * 16;
    const unsigned asBase = lds_addr(As);

    v8f acc[4] = {v8f{}, v8f{}, v8f{}, v8f{}};

    for (int ks = 0; ks < 256; ks += 32) {
        // A tile 128x32 halves: 4 x 16B chunks per row, 512 chunks / 256 thr
#pragma unroll
        for (int j = 0; j < 2; ++j) {
            const int c = tid + 256 * j;
            const int row = c >> 2, ch = (c & 3) * 8;  // 8 halves per chunk
            async_ld16(asBase + (unsigned)(row * 40 + ch) * 2,
                       A + (size_t)(m0 + row) * 256 + ks + ch);
        }
#pragma unroll
        for (int j = 0; j < 2; ++j) {              // W tile f32->f16 (VALU)
            const int c = tid + 256 * j;
            const int row = c >> 3, k4 = (c & 7) << 2;
            const float4 f = *(const float4*)(W + (size_t)(n0 + row) * 256 + ks + k4);
            Bs[row * 40 + k4 + 0] = (_Float16)f.x;
            Bs[row * 40 + k4 + 1] = (_Float16)f.y;
            Bs[row * 40 + k4 + 2] = (_Float16)f.z;
            Bs[row * 40 + k4 + 3] = (_Float16)f.w;
        }
        async_wait0();
        __syncthreads();
        const v16h a = lds_load_a(As, rw, 40, 0, lo, hi);
#pragma unroll
        for (int t = 0; t < 4; ++t) {
            const v16h b = lds_load_b(Bs, 16 * t, 40, 0, lo, hi);
            acc[t] = WMMA_F16(a, b, acc[t]);
        }
        __syncthreads();
    }
#pragma unroll
    for (int t = 0; t < 4; ++t) {
        const int col = n0 + 16 * t + lo;
        const float bcol = bias[col];
#pragma unroll
        for (int r = 0; r < 8; ++r) {
            const int row = m0 + rw + r + 8 * hi;
            Out[(size_t)row * 256 + col] = acc[t][r] + bcol;
        }
    }
}

// ---------------------------------------------------------------------------
// LePE: depthwise 3x3 conv on window_reverse(q), window indexing folded in.
// ---------------------------------------------------------------------------
__global__ void __launch_bounds__(256)
lepe_conv(const _Float16* __restrict__ q, const float* __restrict__ rpe_w,
          const float* __restrict__ rpe_b, _Float16* __restrict__ ao) {
    const size_t g = (size_t)blockIdx.x * 256 + threadIdx.x;
    const int c  = (int)(g & 255);
    const int x0 = (int)((g >> 8) & 255);
    const int y0 = (int)((g >> 16) & 255);
    const int b  = (int)(g >> 24);

    float acc = rpe_b[c];
#pragma unroll
    for (int ky = 0; ky < 3; ++ky) {
        const int yy = y0 + ky - 1;
        if ((unsigned)yy >= 256u) continue;
#pragma unroll
        for (int kx = 0; kx < 3; ++kx) {
            const int xx = x0 + kx - 1;
            if ((unsigned)xx >= 256u) continue;
            const int bw = b * 1024 + (yy >> 3) * 32 + (xx >> 3);
            const int n  = ((yy & 7) << 3) + (xx & 7);
            acc += (float)q[((size_t)(bw * 64 + n) << 8) + c] * rpe_w[c * 9 + ky * 3 + kx];
        }
    }
    const int bo = b * 1024 + (y0 >> 3) * 32 + (x0 >> 3);
    const int no = ((y0 & 7) << 3) + (x0 & 7);
    ao[((size_t)(bo * 64 + no) << 8) + c] = (_Float16)acc;
}

// ---------------------------------------------------------------------------
// Window attention core: one (window b_, head h) per 128-thread workgroup.
// Q/K/V tiles copied raw f16 global->LDS via async ops; S = Q K^T (WMMA),
// epilogue s*scale + mask, row softmax, P f16 -> LDS, O = P V (WMMA),
// ao += O (lepe already resident, f16 RMW with unique ownership).
// ---------------------------------------------------------------------------
__global__ void __launch_bounds__(128)
win_attn_wmma(const _Float16* __restrict__ q, const _Float16* __restrict__ k,
              const _Float16* __restrict__ v, const float* __restrict__ mask,
              _Float16* __restrict__ ao) {
    __shared__ _Float16 Qs[64 * 40];   // [token][d]
    __shared__ _Float16 Ks[64 * 40];   // [token][d]
    __shared__ _Float16 Vr[64 * 40];   // [token][d] (B gathered strided)
    __shared__ _Float16 Ps[64 * 72];   // [row][col]

    const int b_  = blockIdx.x >> 3;
    const int h   = blockIdx.x & 7;
    const int tid = threadIdx.x;
    const int wave = tid >> 5, lane = tid & 31, hi = lane >> 4, lo = lane & 15;
    const int rw = wave * 16;

    const float scale = 0.17677669529663687f;  // 32^-0.5
    const size_t base = ((size_t)b_ * 64) * 256 + (size_t)h * 32;  // halves

    // async stage Q,K,V: 64 tokens x 32 halves = 4 x 16B chunks per token
    const unsigned qsB = lds_addr(Qs), ksB = lds_addr(Ks), vsB = lds_addr(Vr);
#pragma unroll
    for (int j = 0; j < 2; ++j) {
        const int c = tid + 128 * j;               // 0..255
        const int row = c >> 2, ch = (c & 3) * 8;  // 8 halves per chunk
        const unsigned loff = (unsigned)(row * 40 + ch) * 2;
        const size_t goff = base + (size_t)row * 256 + ch;
        async_ld16(qsB + loff, q + goff);
        async_ld16(ksB + loff, k + goff);
        async_ld16(vsB + loff, v + goff);
    }
    async_wait0();
    __syncthreads();

    // S = Q K^T : wave owns rows [rw, rw+16), all 64 cols
    v8f s[4] = {v8f{}, v8f{}, v8f{}, v8f{}};
    {
        const v16h aq = lds_load_a(Qs, rw, 40, 0, lo, hi);
#pragma unroll
        for (int t = 0; t < 4; ++t) {
            const v16h bk = lds_load_b(Ks, 16 * t, 40, 0, lo, hi);
            s[t] = WMMA_F16(aq, bk, s[t]);
        }
    }

    // s*scale + mask (mask shape (nW,64,64); window index = b_ % 1024)
    const float* mrow = mask + ((size_t)(b_ & 1023)) * 4096;
#pragma unroll
    for (int t = 0; t < 4; ++t)
#pragma unroll
        for (int r = 0; r < 8; ++r)
            s[t][r] = fmaf(s[t][r], scale,
                           mrow[(rw + r + 8 * hi) * 64 + 16 * t + lo]);

    // softmax over cols: row r+8*hi lives in one 16-lane half (lo) x 4 tiles
#pragma unroll
    for (int r = 0; r < 8; ++r) {
        float m = fmaxf(fmaxf(s[0][r], s[1][r]), fmaxf(s[2][r], s[3][r]));
#pragma unroll
        for (int off = 8; off >= 1; off >>= 1)
            m = fmaxf(m, __shfl_xor(m, off, 32));
        float sum = 0.0f;
#pragma unroll
        for (int t = 0; t < 4; ++t) {
            const float e = __expf(s[t][r] - m);
            s[t][r] = e;
            sum += e;
        }
#pragma unroll
        for (int off = 8; off >= 1; off >>= 1)
            sum += __shfl_xor(sum, off, 32);
        const float inv = 1.0f / sum;
#pragma unroll
        for (int t = 0; t < 4; ++t)
            Ps[(rw + r + 8 * hi) * 72 + 16 * t + lo] = (_Float16)(s[t][r] * inv);
    }
    __syncthreads();

    // O = P V : K=64 in two 32-steps, N=32 in two 16-tiles
    v8f o[2] = {v8f{}, v8f{}};
#pragma unroll
    for (int kk = 0; kk < 2; ++kk) {
        const v16h ap = lds_load_a(Ps, rw, 72, 32 * kk, lo, hi);
#pragma unroll
        for (int t2 = 0; t2 < 2; ++t2) {
            const v16h bv = lds_load_b_rowmajor(Vr, 16 * t2, 40, 32 * kk, lo, hi);
            o[t2] = WMMA_F16(ap, bv, o[t2]);
        }
    }

    // ao += O  (unique (b_,h) ownership -> plain f16 RMW is safe)
#pragma unroll
    for (int t2 = 0; t2 < 2; ++t2)
#pragma unroll
        for (int r = 0; r < 8; ++r) {
            const size_t idx = base + (size_t)(rw + r + 8 * hi) * 256 + 16 * t2 + lo;
            ao[idx] = (_Float16)((float)ao[idx] + o[t2][r]);
        }
}

// ---------------------------------------------------------------------------
extern "C" void kernel_launch(void* const* d_in, const int* in_sizes, int n_in,
                              void* d_out, int out_size, void* d_ws, size_t ws_size,
                              hipStream_t stream) {
    (void)n_in; (void)out_size; (void)ws_size;
    const float* x     = (const float*)d_in[0];
    const float* y     = (const float*)d_in[1];
    const float* mask  = (const float*)d_in[2];
    const float* Wq    = (const float*)d_in[3];
    const float* bq    = (const float*)d_in[4];
    const float* Wk    = (const float*)d_in[5];
    const float* bk    = (const float*)d_in[6];
    const float* Wv    = (const float*)d_in[7];
    const float* bv    = (const float*)d_in[8];
    const float* rpe_w = (const float*)d_in[9];
    const float* rpe_b = (const float*)d_in[10];
    const float* Wp    = (const float*)d_in[11];
    const float* bp    = (const float*)d_in[12];
    float* out = (float*)d_out;

    const int M = in_sizes[0] / 256;          // 262144 token rows (B_*N)
    const size_t MC = (size_t)M * 256;

    _Float16* q  = (_Float16*)d_ws;           // f16 intermediates (traffic /2)
    _Float16* k  = q + MC;
    _Float16* v  = k + MC;
    _Float16* ao = v + MC;                    // lepe, then += attention out

    const dim3 ggrid(M / 128, 4);             // (2048, 4) 128x64 tiles
    gemm_f32in_f16out<<<ggrid, 256, 0, stream>>>(x, Wq, bq, q);
    gemm_f32in_f16out<<<ggrid, 256, 0, stream>>>(y, Wk, bk, k);
    gemm_f32in_f16out<<<ggrid, 256, 0, stream>>>(x, Wv, bv, v);

    lepe_conv<<<(unsigned)(MC / 256), 256, 0, stream>>>(q, rpe_w, rpe_b, ao);

    win_attn_wmma<<<(unsigned)(M / 64) * 8, 128, 0, stream>>>(q, k, v, mask, ao);

    gemm_f16in_f32out<<<ggrid, 256, 0, stream>>>(ao, Wp, bp, out);
}
